// Attention_16011638079620
// MI455X (gfx1250) — matrix operand order repaired and checked by reading, not compiled, not run on hardware
//
#include <hip/hip_runtime.h>
#include <stdint.h>

typedef _Float16 half_t;
typedef __attribute__((ext_vector_type(16))) _Float16 v16h;
typedef __attribute__((ext_vector_type(8)))  _Float16 v8h;
typedef __attribute__((ext_vector_type(8)))  float    v8f;
typedef __attribute__((ext_vector_type(4)))  float    v4f;

#define NROWS 8192
#define DDIM  256
#define TOPK  10

#define ROWS   64              // rows per block
#define CTILE  64              // cols per tile
#define BSTR   264             // B-tile LDS row stride in halves (256+8 pad -> 528B, 16B aligned, conflict-free)
#define SSTR   65              // sim-tile LDS row stride in floats (odd -> conflict-free column scans)
#define NSPLIT 4               // column splits (parallelism: 128*4 = 512 blocks)
#define COLS_PER_SPLIT (NROWS / NSPLIT)   // 2048
#define NTILE  (COLS_PER_SPLIT / CTILE)   // 32

#define BT_BYTES (ROWS * BSTR * 2)        // 33792 per buffer

#define NEG_INF (-3.4e38f)

// Low 32 bits of a flat LDS address == LDS byte offset (ISA 10.2 aperture layout).
__device__ __forceinline__ unsigned lds_off_u32(const void* p) {
  return (unsigned)(unsigned long long)p;
}

// CDNA5 async global->LDS copy (GVS mode: saddr base + 32-bit vgpr offset), tracked by ASYNCcnt.
__device__ __forceinline__ void async_copy_b128(unsigned lds_byte_off, const void* gbase,
                                                unsigned gbyte_off) {
  asm volatile("global_load_async_to_lds_b128 %0, %1, %2"
               :
               : "v"(lds_byte_off), "v"(gbyte_off), "s"(gbase)
               : "memory");
}
__device__ __forceinline__ void wait_async0() {
  asm volatile("s_wait_asynccnt 0" ::: "memory");
}

// ---------------- Stage 1: row-normalize feats -> f16 unit vectors ----------------
__global__ __launch_bounds__(256) void k_normalize(const float* __restrict__ feats,
                                                   half_t* __restrict__ fhat) {
  int row  = (blockIdx.x * blockDim.x + threadIdx.x) >> 5;  // one wave32 per row
  int lane = threadIdx.x & 31;
  if (row >= NROWS) return;
  const float* rp = feats + (size_t)row * DDIM + lane * 8;
  v4f a = *(const v4f*)(rp);
  v4f b = *(const v4f*)(rp + 4);
  float ss = 0.f;
#pragma unroll
  for (int i = 0; i < 4; ++i) ss += a[i] * a[i] + b[i] * b[i];
#pragma unroll
  for (int m = 16; m >= 1; m >>= 1) ss += __shfl_xor(ss, m, 32);
  float inv = rsqrtf(fmaxf(ss, 1e-24f));
  half_t* out = fhat + (size_t)row * DDIM + lane * 8;
#pragma unroll
  for (int i = 0; i < 4; ++i) out[i] = (half_t)(a[i] * inv);
#pragma unroll
  for (int i = 0; i < 4; ++i) out[4 + i] = (half_t)(b[i] * inv);
}

// ---------------- Stage 2: fused WMMA cosine-sim + streaming top-10 ----------------
// Block = 64 rows x one column split (2048 cols). 8 waves: (wr 0..3 row-strip) x (wc 0..1 col-half).
// Double-buffered async B-tile staging overlaps DMA with WMMA + top-k scan.
__global__ __launch_bounds__(256) void k_simtopk(const half_t* __restrict__ fhat,
                                                 float* __restrict__ pV,
                                                 int* __restrict__ pI) {
  extern __shared__ char smem[];
  // LDS map: [0, BT_BYTES)        = B tile buffer 0
  //          [BT_BYTES, 2*)       = B tile buffer 1
  //          [2*BT_BYTES, ...)    = 64 x SSTR f32 sim tile
  float* simT = (float*)(smem + 2 * BT_BYTES);
  float* mV   = (float*)smem;                              // reuse buffer-0 region at the end
  int*   mI   = (int*)(smem + 256 * TOPK * sizeof(float));

  const int t    = threadIdx.x;
  const int lane = t & 31;
  const int w    = t >> 5;
  const int wr   = w & 3;        // 16-row strip within block
  const int wc   = w >> 2;       // 32-col half of the 64-col tile
  const int hi   = lane >> 4;    // lane half (ISA A/B fragment layout)
  const int ln   = lane & 15;
  const int rowBase  = blockIdx.x * ROWS;
  const int colBase0 = blockIdx.y * COLS_PER_SPLIT;

  // Staging geometry: 64 rows x 512B, 4 threads per row, 8 x b128 per thread.
  const int brow = t >> 2;
  const int bq   = t & 3;
  const unsigned ldsStageBase = lds_off_u32(smem) + (unsigned)(brow * (BSTR * 2) + bq * 128);

  // A strip: 16 rows x 256 K, kept in registers for the whole column sweep.
  // ISA 16-bit A 16x32 layout: lane(lo) halves = K{0..7, 16..23}, lane(hi) = K{8..15, 24..31}.
  union AV { v16h v; v8h h[2]; };
  AV a[8];
  {
    const half_t* arow = fhat + (size_t)(rowBase + wr * 16 + ln) * DDIM;
#pragma unroll
    for (int kk = 0; kk < 8; ++kk) {
      a[kk].h[0] = *(const v8h*)(arow + kk * 32 + hi * 8);
      a[kk].h[1] = *(const v8h*)(arow + kk * 32 + 16 + hi * 8);
    }
  }

  // Per-thread running top-10 (sorted desc). Thread t scans row t&63, col quarter t>>6.
  float tv[TOPK];
  int   ti[TOPK];
#pragma unroll
  for (int s = 0; s < TOPK; ++s) { tv[s] = NEG_INF; ti[s] = 0; }
  const int srow = t & 63;
  const int q    = t >> 6;

  // Prime the pipeline: stage tile 0 into buffer 0.
  {
    unsigned goff = (unsigned)((colBase0 + brow) * (DDIM * 2) + bq * 128);
#pragma unroll
    for (int i = 0; i < 8; ++i) async_copy_b128(ldsStageBase + i * 16, fhat, goff + i * 16);
  }

  for (int ct = 0; ct < NTILE; ++ct) {
    const int colBase = colBase0 + ct * CTILE;
    // Keep buffer selection as pure byte-offset arithmetic on `smem` so clang keeps
    // the addrspace(3) provenance and emits ds_load_b128 (NOT flat_load).
    const unsigned bOffCur = (unsigned)((ct & 1) * BT_BYTES);

    wait_async0();      // my copies into the current buffer are done
    __syncthreads();    // everyone's copies done; everyone finished reading this buffer last round

    // Kick off async staging for the NEXT tile into the other buffer (overlaps compute+scan).
    if (ct + 1 < NTILE) {
      unsigned ldsb = ldsStageBase + (unsigned)(((ct + 1) & 1) * BT_BYTES);
      unsigned goff = (unsigned)((colBase + CTILE + brow) * (DDIM * 2) + bq * 128);
#pragma unroll
      for (int i = 0; i < 8; ++i) async_copy_b128(ldsb + i * 16, fhat, goff + i * 16);
    }

    // WMMA compute: this wave's 16x32 sub-tile of the 64x64 sim tile.
    // Preload all 8 B fragments (16 ds_load_b128, one dscnt wait), then 8 back-to-back WMMAs.
#pragma unroll
    for (int cf = 0; cf < 2; ++cf) {
      const half_t* bcol =
          (const half_t*)(smem + bOffCur) + (wc * 32 + cf * 16 + ln) * BSTR + hi * 8;
      AV b[8];
#pragma unroll
      for (int kk = 0; kk < 8; ++kk) {
        b[kk].h[0] = *(const v8h*)(bcol + kk * 32);
        b[kk].h[1] = *(const v8h*)(bcol + kk * 32 + 16);
      }
      __builtin_amdgcn_sched_barrier(0);   // keep all 16 LDS loads ahead of the WMMA chain
      v8f acc = {};
#pragma unroll
      for (int kk = 0; kk < 8; ++kk) {
        acc = __builtin_amdgcn_wmma_f32_16x16x32_f16(false, a[kk].v, false, b[kk].v,
                                                     (short)0, acc, false, false);
      }
      __builtin_amdgcn_sched_barrier(0);
      // C layout: VGPR r -> row r + 8*hi, col = ln.
#pragma unroll
      for (int r = 0; r < 8; ++r)
        simT[(wr * 16 + r + 8 * hi) * SSTR + wc * 32 + cf * 16 + ln] = acc[r];
    }
    __syncthreads();

    // Threshold-guarded top-10 scan of this tile's 16-column quarter.
#pragma unroll
    for (int c = 0; c < 16; ++c) {
      int   col = q * 16 + c;
      float v   = simT[srow * SSTR + col];
      int   gc  = colBase + col;
      if (v > tv[TOPK - 1]) {
        float cv = v; int ci = gc;
#pragma unroll
        for (int s = 0; s < TOPK; ++s) {
          if (cv > tv[s]) { float fv = tv[s]; int fi = ti[s]; tv[s] = cv; ti[s] = ci; cv = fv; ci = fi; }
        }
      }
    }
    __syncthreads();
  }

  // In-block 4-way merge of the column quarters; emit one sorted list per row per split.
#pragma unroll
  for (int s = 0; s < TOPK; ++s) {
    mV[(srow * 4 + q) * TOPK + s] = tv[s];
    mI[(srow * 4 + q) * TOPK + s] = ti[s];
  }
  __syncthreads();
  if (t < ROWS) {
    int h[4] = {0, 0, 0, 0};
    float* outV = pV + ((size_t)(rowBase + t) * NSPLIT + blockIdx.y) * TOPK;
    int*   outI = pI + ((size_t)(rowBase + t) * NSPLIT + blockIdx.y) * TOPK;
#pragma unroll
    for (int s = 0; s < TOPK; ++s) {
      float best = NEG_INF; int bj = 0;
#pragma unroll
      for (int j = 0; j < 4; ++j) {
        float v = (h[j] < TOPK) ? mV[(t * 4 + j) * TOPK + h[j]] : NEG_INF;
        if (v > best) { best = v; bj = j; }
      }
      int hj = (h[bj] < TOPK) ? h[bj] : (TOPK - 1);
      outV[s] = best;
      outI[s] = mI[(t * 4 + bj) * TOPK + hj];
      h[bj]++;
    }
  }
}

// ---------------- Stage 3: merge splits, degrees, zero-fill, scatter G ----------------
__global__ __launch_bounds__(256) void k_mergetopk(const float* __restrict__ pV,
                                                   const int* __restrict__ pI,
                                                   int* __restrict__ topk) {
  int row = blockIdx.x * blockDim.x + threadIdx.x;
  if (row >= NROWS) return;
  const float* v  = pV + (size_t)row * NSPLIT * TOPK;
  const int*   ix = pI + (size_t)row * NSPLIT * TOPK;
  int h[NSPLIT] = {0, 0, 0, 0};
#pragma unroll
  for (int s = 0; s < TOPK; ++s) {
    float best = NEG_INF; int bj = 0;
#pragma unroll
    for (int j = 0; j < NSPLIT; ++j) {
      float vv = (h[j] < TOPK) ? v[j * TOPK + h[j]] : NEG_INF;
      if (vv > best) { best = vv; bj = j; }
    }
    int hj = (h[bj] < TOPK) ? h[bj] : (TOPK - 1);
    topk[row * TOPK + s] = ix[bj * TOPK + hj];
    h[bj]++;
  }
}

__global__ __launch_bounds__(256) void k_zero_i32(int* __restrict__ p, int n) {
  int i = blockIdx.x * blockDim.x + threadIdx.x;
  if (i < n) p[i] = 0;
}

__global__ __launch_bounds__(256) void k_hist(const int* __restrict__ topk,
                                              int* __restrict__ dv) {
  int i = blockIdx.x * blockDim.x + threadIdx.x;
  if (i < NROWS * TOPK) atomicAdd(&dv[topk[i]], 1);
}

__global__ __launch_bounds__(256) void k_invdv(const int* __restrict__ dv,
                                               float* __restrict__ invdv) {
  int i = blockIdx.x * blockDim.x + threadIdx.x;
  if (i < NROWS) {
    int d = dv[i];
    invdv[i] = (d > 0) ? rsqrtf((float)d) : 0.f;
  }
}

__global__ __launch_bounds__(256) void k_zero_f4(float* __restrict__ p, long long n4) {
  long long i      = (long long)blockIdx.x * blockDim.x + threadIdx.x;
  long long stride = (long long)gridDim.x * blockDim.x;
  v4f z = {};
  for (; i < n4; i += stride) ((v4f*)p)[i] = z;
}

__global__ __launch_bounds__(256) void k_scatter(const int* __restrict__ topk,
                                                 const float* __restrict__ invdv,
                                                 const int* __restrict__ knp,
                                                 float* __restrict__ G) {
  int e = blockIdx.x * blockDim.x + threadIdx.x;
  if (e >= NROWS) return;
  float inv_de = 1.0f / (float)knp[0];
  int   idx[TOPK];
  float wgt[TOPK];
#pragma unroll
  for (int s = 0; s < TOPK; ++s) {
    idx[s] = topk[e * TOPK + s];
    wgt[s] = invdv[idx[s]];
  }
#pragma unroll
  for (int a2 = 0; a2 < TOPK; ++a2)
#pragma unroll
    for (int b2 = 0; b2 < TOPK; ++b2)
      atomicAdd(G + (size_t)idx[a2] * NROWS + idx[b2], inv_de * wgt[a2] * wgt[b2]);
}

extern "C" void kernel_launch(void* const* d_in, const int* in_sizes, int n_in,
                              void* d_out, int out_size, void* d_ws, size_t ws_size,
                              hipStream_t stream) {
  const float* feats = (const float*)d_in[0];
  const int*   kn    = (const int*)d_in[1];
  float*       G     = (float*)d_out;

  char* ws = (char*)d_ws;
  half_t* fhat  = (half_t*)ws;  ws += (size_t)NROWS * DDIM * sizeof(half_t);     // 4 MB
  float*  pV    = (float*)ws;   ws += (size_t)NROWS * NSPLIT * TOPK * sizeof(float);
  int*    pI    = (int*)ws;     ws += (size_t)NROWS * NSPLIT * TOPK * sizeof(int);
  int*    topk  = (int*)ws;     ws += (size_t)NROWS * TOPK * sizeof(int);
  int*    dv    = (int*)ws;     ws += (size_t)NROWS * sizeof(int);
  float*  invdv = (float*)ws;

  k_normalize<<<NROWS / 8, 256, 0, stream>>>(feats, fhat);

  size_t smem = 2 * (size_t)BT_BYTES + (size_t)ROWS * SSTR * sizeof(float); // 84224 B
  dim3 grid(NROWS / ROWS, NSPLIT);  // 128 x 4 = 512 blocks
  k_simtopk<<<grid, 256, smem, stream>>>(fhat, pV, pI);

  k_mergetopk<<<(NROWS + 255) / 256, 256, 0, stream>>>(pV, pI, topk);
  k_zero_i32<<<(NROWS + 255) / 256, 256, 0, stream>>>(dv, NROWS);
  k_hist<<<(NROWS * TOPK + 255) / 256, 256, 0, stream>>>(topk, dv);
  k_invdv<<<(NROWS + 255) / 256, 256, 0, stream>>>(dv, invdv);
  k_zero_f4<<<4096, 256, 0, stream>>>(G, (long long)NROWS * NROWS / 4);
  k_scatter<<<(NROWS + 255) / 256, 256, 0, stream>>>(topk, invdv, kn, G);
}
